// BoltzmannModel_87093346828726
// MI455X (gfx1250) — matrix operand order repaired and checked by reading, not compiled, not run on hardware
//
#include <hip/hip_runtime.h>
#include <math.h>

typedef __attribute__((ext_vector_type(16))) _Float16 v16h;
typedef __attribute__((ext_vector_type(8)))  float    v8f;

#define DIM  26
#define SKEW 10.0f

// Workspace layout (d_ws):
//  u32[0] enc max_all   u32[1] enc max_wrong   f32[2] sum_all  f32[3] sum_wrong
//  i32[4] correct out   i32[5] input bits
//  byte   64 : WhiT _Float16[32*32] (2048 B)  TRANSPOSED: WhiT[col*32 + k]
//  byte 2112 : WloT _Float16[32*32] (2048 B)  (k rows 0..17 cross, 18 eps|delta)
//  byte 4160 : GamT float[16*16]    (1024 B)  TRANSPOSED: GamT[n*16 + m]
//  byte 5184 : K    float[1024]     (4096 B)

__device__ __forceinline__ unsigned encf(float f) {
  unsigned u = __float_as_uint(f);
  return (u & 0x80000000u) ? ~u : (u | 0x80000000u);
}
__device__ __forceinline__ float decf(unsigned m) {
  return (m & 0x80000000u) ? __uint_as_float(m & 0x7fffffffu)
                           : __uint_as_float(~m);
}
__device__ __forceinline__ int pairCoeff(int i, int j) {  // i<j, coeffs index
  return DIM + (i * (2 * DIM - i - 1)) / 2 + (j - i - 1);
}

__global__ void boltz_setup(const int* input_int, const int* answer_table,
                            const float* coeffs, void* ws) {
  unsigned* wsu = (unsigned*)ws;
  int*      wsi = (int*)ws;
  float*    wsf = (float*)ws;
  _Float16* whiT = (_Float16*)((char*)ws + 64);
  _Float16* wloT = whiT + 1024;
  float*    gamT = (float*)((char*)ws + 4160);
  int t = threadIdx.x;
  if (t == 0) {
    wsu[0] = 0x007FFFFFu;  // enc(-inf)
    wsu[1] = 0x007FFFFFu;
    wsf[2] = 0.0f;
    wsf[3] = 0.0f;
    wsi[4] = answer_table[input_int[0]];
    wsi[5] = input_int[0] & 0xFF;
  }
  // W[k][col]: k<18: cross terms t-dim k -> p cols (n<16) / q cols (16+m);
  //            k==18: eps(n) | delta(m); k=19..31: 0.  Stored TRANSPOSED.
  for (int idx = t; idx < 1024; idx += blockDim.x) {
    int k = idx >> 5, col = idx & 31;
    float v = 0.0f;
    int base = (col < 16) ? 18 : 22;
    int bits = (col < 16) ? col : (col - 16);
    if (k < 18) {
      for (int j = 0; j < 4; ++j)
        if ((bits >> j) & 1) v += coeffs[pairCoeff(k, base + j)];
    } else if (k == 18) {
      for (int a = 0; a < 4; ++a)
        if ((bits >> a) & 1) {
          v += coeffs[base + a];  // linear term
          for (int b = a + 1; b < 4; ++b)
            if ((bits >> b) & 1) v += coeffs[pairCoeff(base + a, base + b)];
        }
    }
    _Float16 hi = (_Float16)v;
    whiT[col * 32 + k] = hi;
    wloT[col * 32 + k] = (_Float16)(v - (float)hi);
  }
  // GamT[n*16 + m] = sum_{i in 18..21, j in 22..25} c(i,j)*bit(n,i-18)*bit(m,j-22)
  if (t < 256) {
    int m = t & 15, n = t >> 4;
    float v = 0.0f;
    for (int i = 0; i < 4; ++i)
      if ((n >> i) & 1)
        for (int j = 0; j < 4; ++j)
          if ((m >> j) & 1) v += coeffs[pairCoeff(18 + i, 22 + j)];
    gamT[t] = v;
  }
}

// K_t: quadratic form over the 18 fixed dims (input bits + out bits of t).
__global__ void boltz_kconst(const int* input_int, const float* coeffs, void* ws) {
  int t = blockIdx.x * blockDim.x + threadIdx.x;  // 0..1023
  unsigned mk = ((unsigned)input_int[0] & 0xFFu) | ((unsigned)t << 8);
  float acc = 0.0f;
  for (int i = 0; i < 18; ++i) {
    if ((mk >> i) & 1u) {
      acc += coeffs[i];
      for (int j = i + 1; j < 18; ++j)
        if ((mk >> j) & 1u) acc += coeffs[pairCoeff(i, j)];
    }
  }
  float* karr = (float*)((char*)ws + 5184);
  karr[t] = acc;
}

// PASS 0: maxima of log-factors (all/wrong).  PASS 1: shifted exp sums.
// One wave = one WMMA tile of 16 supertiles (out values) = 4096 states.
template <int PASS>
__global__ __launch_bounds__(256) void boltz_pass(void* ws) {
  const unsigned*  wsu  = (const unsigned*)ws;
  float*           wsf  = (float*)ws;
  const int*       wsi  = (const int*)ws;
  const _Float16*  whiT = (const _Float16*)((const char*)ws + 64);
  const _Float16*  wloT = whiT + 1024;
  const float*     gamT = (const float*)((const char*)ws + 4160);
  const float*     karr = (const float*)((const char*)ws + 5184);

  __shared__ float XT[8][512];   // per-wave 16x32 GEMM result
  __shared__ float red[256];

  const int tid  = threadIdx.x;
  const int lane = tid & 31;
  const int w    = tid >> 5;
  const int h    = lane >> 4;
  const int n    = lane & 15;
  const int correct = wsi[4];
  const unsigned inbits = (unsigned)wsi[5];

  float Ma = 0.0f, Mw = 0.0f;
  if (PASS == 1) { Ma = decf(wsu[0]); Mw = decf(wsu[1]); }

  // B fragments (KxN): lane -> N = lane&15, K = 16h + element.
  // Transposed storage => 16 contiguous halves per fragment (2x b128 each).
  v16h bh0 = *(const v16h*)(whiT + n * 32 + 16 * h);
  v16h bl0 = *(const v16h*)(wloT + n * 32 + 16 * h);
  v16h bh1 = *(const v16h*)(whiT + (n + 16) * 32 + 16 * h);
  v16h bl1 = *(const v16h*)(wloT + (n + 16) * 32 + 16 * h);

  // A fragment (MxK): row = supertile (out id), K = [t bits(18), 1, 0...].
  const int tileBase = (blockIdx.x * 8 + w) * 16;   // 64 waves x 16 = 1024
  v16h a;
  {
    unsigned mk = inbits | ((unsigned)(tileBase + n) << 8) | (1u << 18);
    int kb = 16 * h;
#pragma unroll
    for (int e = 0; e < 8; ++e)
      a[e] = (_Float16)(float)((mk >> (kb + e)) & 1u);
#pragma unroll
    for (int e = 8; e < 16; ++e)
      a[e] = (_Float16)(float)((mk >> (kb + e)) & 1u);
  }

  // X = A * (Wlo + Whi): cols 0..15 -> R_t[n], cols 16..31 -> C_t[m].
  v8f c0 = {}; v8f c1 = {};
  c0 = __builtin_amdgcn_wmma_f32_16x16x32_f16(false, a, false, bl0, (short)0, c0, false, false);
  c0 = __builtin_amdgcn_wmma_f32_16x16x32_f16(false, a, false, bh0, (short)0, c0, false, false);
  c1 = __builtin_amdgcn_wmma_f32_16x16x32_f16(false, a, false, bl1, (short)0, c1, false, false);
  c1 = __builtin_amdgcn_wmma_f32_16x16x32_f16(false, a, false, bh1, (short)0, c1, false, false);

  // Stash D fragments to LDS once (batched stores, single barrier).
#pragma unroll
  for (int r = 0; r < 8; ++r) {
    XT[w][(r + 8 * h) * 32 + n]      = c0[r];
    XT[w][(r + 8 * h) * 32 + 16 + n] = c1[r];
  }
  __syncthreads();

  // Gamma column for this lane's n: 8 contiguous floats (2x b128).
  float g[8];
  {
    const float* gp = gamT + n * 16 + 8 * h;
#pragma unroll
    for (int r = 0; r < 8; ++r) g[r] = gp[r];
  }

  float accMaxAll = -INFINITY, accMaxWrong = -INFINITY;
  float accSumAll = 0.0f, accSumWrong = 0.0f;

  for (int tl = 0; tl < 16; ++tl) {
    int t = tileBase + tl;
    float base = karr[t] + XT[w][tl * 32 + n];         // K_t + R_t[n]
    const float* cbase = &XT[w][tl * 32 + 16 + 8 * h]; // C_t[8h..8h+7], b128-able
    bool wrongT = (t != correct);                      // wave-uniform
#pragma unroll
    for (int r = 0; r < 8; ++r) {
      float lf = -(g[r] + cbase[r] + base);
      if (PASS == 0) {
        accMaxAll = fmaxf(accMaxAll, lf);
        if (wrongT) accMaxWrong = fmaxf(accMaxWrong, lf);
      } else {
        accSumAll += __expf(lf - Ma);
        if (wrongT) accSumWrong += __expf(lf - Mw);
      }
    }
  }

  // Block reductions + global atomics.
  red[tid] = (PASS == 0) ? accMaxAll : accSumAll;
  __syncthreads();
  for (int off = 128; off > 0; off >>= 1) {
    if (tid < off)
      red[tid] = (PASS == 0) ? fmaxf(red[tid], red[tid + off])
                             : (red[tid] + red[tid + off]);
    __syncthreads();
  }
  if (tid == 0) {
    if (PASS == 0) atomicMax((unsigned*)ws + 0, encf(red[0]));
    else           atomicAdd(wsf + 2, red[0]);
  }
  __syncthreads();
  red[tid] = (PASS == 0) ? accMaxWrong : accSumWrong;
  __syncthreads();
  for (int off = 128; off > 0; off >>= 1) {
    if (tid < off)
      red[tid] = (PASS == 0) ? fmaxf(red[tid], red[tid + off])
                             : (red[tid] + red[tid + off]);
    __syncthreads();
  }
  if (tid == 0) {
    if (PASS == 0) atomicMax((unsigned*)ws + 1, encf(red[0]));
    else           atomicAdd(wsf + 3, red[0]);
  }
}

__global__ void boltz_final(void* ws, float* out) {
  const unsigned* wsu = (const unsigned*)ws;
  const float*    wsf = (const float*)ws;
  float lseA = decf(wsu[0]) + logf(wsf[2]);
  float lseW = decf(wsu[1]) + logf(wsf[3]);
  out[0] = expf(SKEW * (lseW - lseA));
}

extern "C" void kernel_launch(void* const* d_in, const int* in_sizes, int n_in,
                              void* d_out, int out_size, void* d_ws, size_t ws_size,
                              hipStream_t stream) {
  const int*   input_int    = (const int*)d_in[0];
  const int*   answer_table = (const int*)d_in[1];
  const float* coeffs       = (const float*)d_in[2];
  float*       out          = (float*)d_out;

  boltz_setup<<<1, 256, 0, stream>>>(input_int, answer_table, coeffs, d_ws);
  boltz_kconst<<<4, 256, 0, stream>>>(input_int, coeffs, d_ws);
  boltz_pass<0><<<8, 256, 0, stream>>>(d_ws);  // maxima
  boltz_pass<1><<<8, 256, 0, stream>>>(d_ws);  // shifted exp sums
  boltz_final<<<1, 1, 0, stream>>>(d_ws, out);
}
